// MultiHeadAttention_26929444946351
// MI455X (gfx1250) — matrix operand order repaired and checked by reading, not compiled, not run on hardware
//
#include <hip/hip_runtime.h>
#include <hip/hip_bf16.h>
#include <math.h>

// Problem constants (match the reference).
#define B_ 4
#define S_ 4096
#define D_ 512

typedef __attribute__((ext_vector_type(16))) __bf16 v16bf;
typedef __attribute__((ext_vector_type(8)))  __bf16 v8bf;
typedef __attribute__((ext_vector_type(8)))  float  v8f;

static __device__ __forceinline__ __bf16 f2bf(float f) { return (__bf16)f; }

static __device__ __forceinline__ v8f wmma_bf16(v16bf a, v16bf b, v8f c) {
  // D = A(16x32 bf16) x B(32x16 bf16) + C(16x16 f32)
  return __builtin_amdgcn_wmma_f32_16x16x32_bf16(false, a, false, b, (short)0, c,
                                                 false, false);
}

// Two 8-element bf16 chunks (p..p+7, p+16..p+23) -> B fragment (table k order).
static __device__ __forceinline__ v16bf load16(const __bf16* p) {
  v8bf lo = *reinterpret_cast<const v8bf*>(p);
  v8bf hi = *reinterpret_cast<const v8bf*>(p + 16);
  v16bf r;
#pragma unroll
  for (int i = 0; i < 8; ++i) { r[i] = lo[i]; r[i + 8] = hi[i]; }
  return r;
}

// A fragment from its two 8-element chunks (k = 8h..8h+7 and 16+8h..16+8h+7).
static __device__ __forceinline__ v16bf packA(v8bf lo, v8bf hi) {
  v16bf r;
#pragma unroll
  for (int i = 0; i < 8; ++i) { r[i] = lo[i]; r[i + 8] = hi[i]; }
  return r;
}

// ---------------------------------------------------------------------------
// Kernel 1: transpose+convert wq/wk/wv (f32 [k][n]) -> bf16 wT[which][n][k]
// ---------------------------------------------------------------------------
__global__ void pack_w_kernel(const float* __restrict__ wq,
                              const float* __restrict__ wk,
                              const float* __restrict__ wv,
                              __bf16* __restrict__ wT) {
  const int idx = blockIdx.x * blockDim.x + threadIdx.x;
  if (idx >= 3 * D_ * D_) return;
  const int which = idx / (D_ * D_);
  const int rem = idx - which * (D_ * D_);
  const int n = rem >> 9;          // output row (original column)
  const int k = rem & (D_ - 1);    // output col (original row)
  const float* w = (which == 0) ? wq : (which == 1) ? wk : wv;
  wT[idx] = f2bf(w[(size_t)k * D_ + n]);
}

// ---------------------------------------------------------------------------
// Kernel 2: QKV projection GEMM. One wave -> one 16x16 output tile.
//   Q: bf16 [b*s][d], pre-scaled by 1/sqrt(d_k)=0.125
//   K: bf16 [b*s][d]
//   V: bf16 transposed [b][d][t]  (so PV B-fragments are contiguous in t)
// ---------------------------------------------------------------------------
__global__ __launch_bounds__(256) void qkv_kernel(const float* __restrict__ x,
                                                  const __bf16* __restrict__ wT,
                                                  __bf16* __restrict__ Qbf,
                                                  __bf16* __restrict__ Kbf,
                                                  __bf16* __restrict__ Vt) {
  const int lane = threadIdx.x & 31;
  const int wave = threadIdx.x >> 5;
  const int m = lane & 15;  // A row / B column within the tile
  const int g = lane >> 4;  // lane half
  const int ntile = blockIdx.x * 8 + wave;  // 0..31
  const int mtile = blockIdx.y;             // 0..1023
  const int which = blockIdx.z;             // 0=Q 1=K 2=V

  const float* arow = x + ((size_t)mtile * 16 + m) * D_;
  const __bf16* brow =
      wT + (size_t)which * D_ * D_ + ((size_t)ntile * 16 + m) * D_ + 8 * g;

  v8f c;
#pragma unroll
  for (int i = 0; i < 8; ++i) c[i] = 0.0f;

#pragma unroll
  for (int kc = 0; kc < 16; ++kc) {
    const int k0 = kc * 32;
    const float4 a0 = *(const float4*)(arow + k0 + 8 * g);
    const float4 a1 = *(const float4*)(arow + k0 + 8 * g + 4);
    const float4 a2 = *(const float4*)(arow + k0 + 16 + 8 * g);
    const float4 a3 = *(const float4*)(arow + k0 + 16 + 8 * g + 4);
    v16bf a;
    a[0] = f2bf(a0.x);  a[1] = f2bf(a0.y);  a[2] = f2bf(a0.z);  a[3] = f2bf(a0.w);
    a[4] = f2bf(a1.x);  a[5] = f2bf(a1.y);  a[6] = f2bf(a1.z);  a[7] = f2bf(a1.w);
    a[8] = f2bf(a2.x);  a[9] = f2bf(a2.y);  a[10] = f2bf(a2.z); a[11] = f2bf(a2.w);
    a[12] = f2bf(a3.x); a[13] = f2bf(a3.y); a[14] = f2bf(a3.z); a[15] = f2bf(a3.w);
    c = wmma_bf16(a, load16(brow + k0), c);
  }

  const int col = ntile * 16 + m;
#pragma unroll
  for (int r = 0; r < 8; ++r) {
    const int row = mtile * 16 + r + 8 * g;  // flat B*S row
    const float v = c[r];
    if (which == 0) {
      Qbf[(size_t)row * D_ + col] = f2bf(v * 0.125f);  // fold 1/SCALE into Q
    } else if (which == 1) {
      Kbf[(size_t)row * D_ + col] = f2bf(v);
    } else {
      const int bb = row >> 12;       // /4096
      const int t = row & (S_ - 1);
      Vt[((size_t)bb * D_ + col) * S_ + t] = f2bf(v);
    }
  }
}

// ---------------------------------------------------------------------------
// Kernel 3: flash attention (one pass over keys) + residual + LayerNorm.
// One wave owns 16 query rows and the full 16x512 f32 accumulator.
// ---------------------------------------------------------------------------
__global__ __launch_bounds__(128, 1) void attn_ln_kernel(
    const float* __restrict__ x, const float* __restrict__ ln_g,
    const float* __restrict__ ln_b, const __bf16* __restrict__ Qbf,
    const __bf16* __restrict__ Kbf, const __bf16* __restrict__ Vt,
    float* __restrict__ out) {
  __shared__ __bf16 pst[4][16][32];  // per-wave P staging (C-layout -> A-layout)

  const int lane = threadIdx.x & 31;
  const int wave = threadIdx.x >> 5;
  const int m = lane & 15;
  const int g = lane >> 4;
  const int b = blockIdx.y;
  const int q0 = (blockIdx.x * 4 + wave) * 16;  // query row block within batch

  // Q A-fragments for all 16 d-chunks, kept resident in VGPRs (128 regs).
  const __bf16* Qrow = Qbf + ((size_t)b * S_ + q0 + m) * D_;
  v16bf qf[16];
#pragma unroll
  for (int c = 0; c < 16; ++c)
    qf[c] = packA(*(const v8bf*)(Qrow + c * 32 + 8 * g),
                  *(const v8bf*)(Qrow + c * 32 + 16 + 8 * g));

  // Output accumulator: 32 d-tiles of 16x16 f32 (256 VGPRs).
  v8f o[32];
#pragma unroll
  for (int dt = 0; dt < 32; ++dt)
#pragma unroll
    for (int r = 0; r < 8; ++r) o[dt][r] = 0.0f;

  float rm[8], rl[8];  // running row max / row sum (row = r + 8*g)
#pragma unroll
  for (int r = 0; r < 8; ++r) { rm[r] = -INFINITY; rl[r] = 0.0f; }

  const __bf16* Kb = Kbf + (size_t)b * S_ * D_;
  const __bf16* Vb = Vt + (size_t)b * D_ * S_;

  for (int t0 = 0; t0 < S_; t0 += 32) {
    // ---- scores S = Q K^T for 32 keys (two 16x16 n-tiles) ----
    v8f s0, s1;
#pragma unroll
    for (int i = 0; i < 8; ++i) { s0[i] = 0.0f; s1[i] = 0.0f; }
    const __bf16* k0p = Kb + (size_t)(t0 + m) * D_ + 8 * g;  // key row for n-tile0
    const __bf16* k1p = k0p + (size_t)16 * D_;
#pragma unroll
    for (int c = 0; c < 16; ++c) {
      s0 = wmma_bf16(qf[c], load16(k0p + c * 32), s0);
      s1 = wmma_bf16(qf[c], load16(k1p + c * 32), s1);
    }

    // ---- online softmax: row stats via shuffles within 16-lane row groups ----
    float alpha[8];
#pragma unroll
    for (int r = 0; r < 8; ++r) {
      float mx = fmaxf(s0[r], s1[r]);
#pragma unroll
      for (int msk = 1; msk < 16; msk <<= 1) mx = fmaxf(mx, __shfl_xor(mx, msk, 32));
      const float mn = fmaxf(rm[r], mx);
      alpha[r] = __expf(rm[r] - mn);  // 0 on first tile (exp(-inf))
      rm[r] = mn;
      const float p0 = __expf(s0[r] - mn);
      const float p1 = __expf(s1[r] - mn);
      float ps = p0 + p1;
#pragma unroll
      for (int msk = 1; msk < 16; msk <<= 1) ps += __shfl_xor(ps, msk, 32);
      rl[r] = rl[r] * alpha[r] + ps;
      const int row = r + 8 * g;
      pst[wave][row][m] = f2bf(p0);
      pst[wave][row][16 + m] = f2bf(p1);
    }

    // Rescale running output by alpha (per-row).
#pragma unroll
    for (int dt = 0; dt < 32; ++dt)
#pragma unroll
      for (int r = 0; r < 8; ++r) o[dt][r] *= alpha[r];

    __syncthreads();  // LDS stage visible (uniform trip count across waves)

    // Re-shape P into an A fragment (16x32 bf16) from LDS.
    const v16bf pf = packA(*(const v8bf*)&pst[wave][m][8 * g],
                           *(const v8bf*)&pst[wave][m][16 + 8 * g]);

    // ---- O += P V : 32 d-tiles, B fragments contiguous in t from Vt ----
#pragma unroll
    for (int dt = 0; dt < 32; ++dt) {
      const __bf16* vp = Vb + (size_t)(dt * 16 + m) * S_ + t0 + 8 * g;
      o[dt] = wmma_bf16(pf, load16(vp), o[dt]);
    }
  }

  // ---- epilogue: normalize, residual, LayerNorm, store f32 ----
  float inv[8];
#pragma unroll
  for (int r = 0; r < 8; ++r) inv[r] = 1.0f / rl[r];

  const float* xb = x + ((size_t)b * S_ + q0) * D_;

  float mean[8];
#pragma unroll
  for (int r = 0; r < 8; ++r) mean[r] = 0.0f;
#pragma unroll
  for (int dt = 0; dt < 32; ++dt)
#pragma unroll
    for (int r = 0; r < 8; ++r) {
      const float val =
          o[dt][r] * inv[r] + xb[(size_t)(r + 8 * g) * D_ + dt * 16 + m];
      o[dt][r] = val;
      mean[r] += val;
    }
#pragma unroll
  for (int r = 0; r < 8; ++r) {
#pragma unroll
    for (int msk = 1; msk < 16; msk <<= 1) mean[r] += __shfl_xor(mean[r], msk, 32);
    mean[r] *= (1.0f / D_);
  }

  float var[8];
#pragma unroll
  for (int r = 0; r < 8; ++r) var[r] = 0.0f;
#pragma unroll
  for (int dt = 0; dt < 32; ++dt)
#pragma unroll
    for (int r = 0; r < 8; ++r) {
      const float d = o[dt][r] - mean[r];
      var[r] += d * d;
    }
  float rstd[8];
#pragma unroll
  for (int r = 0; r < 8; ++r) {
#pragma unroll
    for (int msk = 1; msk < 16; msk <<= 1) var[r] += __shfl_xor(var[r], msk, 32);
    rstd[r] = rsqrtf(var[r] * (1.0f / D_) + 1e-5f);
  }

#pragma unroll
  for (int dt = 0; dt < 32; ++dt) {
    const int d = dt * 16 + m;
    const float gv = ln_g[d];
    const float bv = ln_b[d];
#pragma unroll
    for (int r = 0; r < 8; ++r) {
      const size_t row = (size_t)b * S_ + q0 + r + 8 * g;
      out[row * D_ + d] = (o[dt][r] - mean[r]) * rstd[r] * gv + bv;
    }
  }
}

// ---------------------------------------------------------------------------
// Launcher. Workspace layout (bf16):
//   wT : 3*512*512            (1.5 MiB)
//   Q  : 4*4096*512           (16 MiB)
//   K  : 4*4096*512           (16 MiB)
//   Vt : 4*512*4096           (16 MiB)   total 49.5 MiB
// ---------------------------------------------------------------------------
extern "C" void kernel_launch(void* const* d_in, const int* in_sizes, int n_in,
                              void* d_out, int out_size, void* d_ws,
                              size_t ws_size, hipStream_t stream) {
  const float* x = (const float*)d_in[0];
  const float* wq = (const float*)d_in[1];
  const float* wk = (const float*)d_in[2];
  const float* wv = (const float*)d_in[3];
  const float* ln_g = (const float*)d_in[4];
  const float* ln_b = (const float*)d_in[5];

  char* ws = (char*)d_ws;
  const size_t wT_bytes = (size_t)3 * D_ * D_ * 2;
  const size_t mat_bytes = (size_t)B_ * S_ * D_ * 2;
  __bf16* wT = (__bf16*)ws;
  __bf16* Qbf = (__bf16*)(ws + wT_bytes);
  __bf16* Kbf = (__bf16*)(ws + wT_bytes + mat_bytes);
  __bf16* Vt = (__bf16*)(ws + wT_bytes + 2 * mat_bytes);

  pack_w_kernel<<<(3 * D_ * D_ + 255) / 256, 256, 0, stream>>>(wq, wk, wv, wT);

  // 3 matrices x (1024 M-tiles) x (32 N-tiles, 8 per block)
  qkv_kernel<<<dim3(D_ / 16 / 8, (B_ * S_) / 16, 3), 256, 0, stream>>>(
      x, wT, Qbf, Kbf, Vt);

  // 4 waves per block, each owning a 16-row query tile.
  attn_ln_kernel<<<dim3(S_ / 64, B_), 128, 0, stream>>>(x, ln_g, ln_b, Qbf, Kbf,
                                                        Vt, (float*)d_out);
}
